// MultiHeadedAttention_90941637526101
// MI455X (gfx1250) — matrix operand order repaired and checked by reading, not compiled, not run on hardware
//
#include <hip/hip_runtime.h>
#include <hip/hip_bf16.h>
#include <cstdint>

typedef __attribute__((ext_vector_type(16))) _Float16 v16h;
typedef __attribute__((ext_vector_type(8)))  float    v8f;
typedef __attribute__((ext_vector_type(2)))  float    f32x2;

#define BATCH 8
#define SLEN 1024
#define DMODEL 512
#define NHEAD 8
#define DK 64

#define ASTR 40   // LDS row stride (f16 elems) for A tiles: 80B rows -> 16B-aligned b128 slots
#define BSTR 72   // LDS row stride for B tiles (mult of 8 -> 16B-aligned frag loads)
#define PSTR 36   // per-wave P tile stride in attention

// A-fragment (16x32 f16) per-lane K pair offsets: VGPR j holds K = {0,2,4,6,16,18,20,22}+8*half
__device__ __forceinline__ int apair(int j, int half) {
  return ((j < 4) ? (2 * j) : (16 + 2 * (j - 4))) + 8 * half;
}

union AFrag { v16h v; uint32_t u[8]; };
union BFrag { v16h v; uint4 q[2]; };
union PackH { uint32_t u; _Float16 h[2]; };

// ---------------------------------------------------------------------------
// Kernel 1: Q/K/V projections.  out = X @ W^T, f32 in -> f16 out.
// z = blockIdx.z selects (X, W, output layout).  K-projection stored transposed
// as Kt[b,h,d,s] so attention B-fragments are contiguous.
// ---------------------------------------------------------------------------
__global__ __launch_bounds__(256) void proj_qkv_kernel(
    const float* __restrict__ Xq, const float* __restrict__ Xk, const float* __restrict__ Xv,
    const float* __restrict__ Wq, const float* __restrict__ Wk, const float* __restrict__ Wv,
    _Float16* __restrict__ Qp, _Float16* __restrict__ Kt, _Float16* __restrict__ Vp)
{
  __shared__ _Float16 As[128 * ASTR];
  __shared__ _Float16 Bs[32 * BSTR];
  const int z = blockIdx.z;
  const float* __restrict__ X = (z == 0) ? Xq : (z == 1) ? Xk : Xv;
  const float* __restrict__ W = (z == 0) ? Wq : (z == 1) ? Wk : Wv;

  const int tid  = threadIdx.x;
  const int lane = tid & 31;
  const int wave = tid >> 5;
  const int half = lane >> 4;
  const int ln   = lane & 15;
  const int wr   = wave & 3;   // 4 row-waves
  const int wc   = wave >> 2;  // 2 col-waves
  const int row0 = blockIdx.x * 128;
  const int col0 = blockIdx.y * 64;

  v8f acc[2][2] = {};

  for (int kc = 0; kc < DMODEL; kc += 32) {
    // stage A: 128x32, f32 -> f16, packed dword stores (8 x b32 per thread)
#pragma unroll
    for (int i = 0; i < 8; ++i) {
      int idx = tid + i * 256;            // dword-pair index 0..2047
      int r = idx >> 4, c2 = (idx & 15) * 2;
      f32x2 xv = *(const f32x2*)(const void*)&X[(size_t)(row0 + r) * DMODEL + kc + c2];
      PackH pk;
      pk.h[0] = (_Float16)xv.x;
      pk.h[1] = (_Float16)xv.y;
      *(uint32_t*)(void*)&As[r * ASTR + c2] = pk.u;
    }
    // stage B transposed: Bs[c][n] = W[col0+n][kc+c], packed n-pairs -> b32 stores
#pragma unroll
    for (int i = 0; i < 4; ++i) {
      int idx = tid + i * 256;            // n-pair index 0..1023
      int np = idx >> 5, c = idx & 31;
      int n = np * 2;
      PackH pk;
      pk.h[0] = (_Float16)W[(size_t)(col0 + n)     * DMODEL + kc + c];
      pk.h[1] = (_Float16)W[(size_t)(col0 + n + 1) * DMODEL + kc + c];
      *(uint32_t*)(void*)&Bs[c * BSTR + n] = pk.u;
    }
    __syncthreads();

    AFrag a[2];
    BFrag bf[2];
#pragma unroll
    for (int i = 0; i < 2; ++i) {
      int m = wr * 32 + i * 16 + ln;
#pragma unroll
      for (int j = 0; j < 8; ++j) {
        int ko = apair(j, half);
        a[i].u[j] = *(const uint32_t*)(const void*)&As[m * ASTR + ko];
      }
    }
#pragma unroll
    for (int j = 0; j < 2; ++j) {
#pragma unroll
      for (int e = 0; e < 16; ++e)
        bf[j].v[e] = Bs[(apair(e >> 1, half) + (e & 1)) * BSTR + wc * 32 + j * 16 + ln];
    }
#pragma unroll
    for (int i = 0; i < 2; ++i)
#pragma unroll
      for (int j = 0; j < 2; ++j)
        acc[i][j] = __builtin_amdgcn_wmma_f32_16x16x32_f16(
            false, a[i].v, false, bf[j].v, (short)0, acc[i][j], false, false);
    __syncthreads();
  }

  // store: split-head layouts
#pragma unroll
  for (int i = 0; i < 2; ++i) {
#pragma unroll
    for (int j = 0; j < 2; ++j) {
#pragma unroll
      for (int r = 0; r < 8; ++r) {
        int row = row0 + wr * 32 + i * 16 + r + 8 * half;
        int col = col0 + wc * 32 + j * 16 + ln;
        int b = row >> 10, s = row & 1023;
        int h = col >> 6,  d = col & 63;
        _Float16 val = (_Float16)acc[i][j][r];
        if (z == 1) {
          Kt[((size_t)(b * NHEAD + h) * DK + d) * SLEN + s] = val;
        } else if (z == 0) {
          Qp[((size_t)(b * NHEAD + h) * SLEN + s) * DK + d] = val;
        } else {
          Vp[((size_t)(b * NHEAD + h) * SLEN + s) * DK + d] = val;
        }
      }
    }
  }
}

// ---------------------------------------------------------------------------
// Kernel 2: flash attention with mask|eye and post-softmax group_prob scaling.
// One 16-query tile per wave; 32 keys per step.  Wave decode puts the 8 heads
// of one (b, q-tile) in one workgroup so mask/group_prob addresses are
// identical across waves -> WGP$/L2 hits instead of 8x HBM reads.
// ---------------------------------------------------------------------------
__global__ __launch_bounds__(256) void attn_kernel(
    const _Float16* __restrict__ Qp, const _Float16* __restrict__ Kt,
    const _Float16* __restrict__ Vp, const float* __restrict__ Gp,
    const int* __restrict__ Mask, _Float16* __restrict__ Xo)
{
  __shared__ _Float16 Plds[8 * 16 * PSTR];
  const int tid  = threadIdx.x;
  const int lane = tid & 31;
  const int wave = tid >> 5;
  const int half = lane >> 4;
  const int ln   = lane & 15;
  const int gw   = blockIdx.x * 8 + wave;
  const int h    = gw & 7;             // head fastest -> shared mask/gp per block
  const int rest = gw >> 3;
  const int qt   = rest & 63;          // S/16 = 64 query tiles
  const int b    = rest >> 6;
  const int bh   = b * NHEAD + h;
  const int q0   = qt * 16;

  _Float16* Pw = &Plds[wave * 16 * PSTR];

  // Q A-fragments (d = 0..31, 32..63), loaded once
  AFrag aq[2];
  const _Float16* qbase = Qp + ((size_t)bh * SLEN + q0) * DK;
#pragma unroll
  for (int f = 0; f < 2; ++f)
#pragma unroll
    for (int j = 0; j < 8; ++j) {
      int ko = apair(j, half) + f * 32;
      aq[f].u[j] = *(const uint32_t*)(const void*)(qbase + (size_t)ln * DK + ko);
    }

  v8f acc[4] = {};
  float mrun[8], lsum[8];
#pragma unroll
  for (int r = 0; r < 8; ++r) { mrun[r] = -3.0e38f; lsum[r] = 0.0f; }

  const float*    gpRow  = Gp   + (size_t)b * SLEN * SLEN;
  const int*      mkRow  = Mask + (size_t)b * SLEN * SLEN;
  const _Float16* ktBase = Kt + (size_t)bh * DK * SLEN;
  const _Float16* vBase  = Vp + (size_t)bh * SLEN * DK;

  for (int s0 = 0; s0 < SLEN; s0 += 32) {
    // ---- scores: two 16x16 tiles via WMMA over d (K=32 x2) ----
    v8f sc[2];
#pragma unroll
    for (int jj = 0; jj < 2; ++jj) {
      v8f s = {};
#pragma unroll
      for (int f = 0; f < 2; ++f) {
        BFrag bk;
#pragma unroll
        for (int e = 0; e < 16; ++e)
          bk.v[e] = ktBase[(size_t)(f * 32 + apair(e >> 1, half) + (e & 1)) * SLEN + s0 + jj * 16 + ln];
        s = __builtin_amdgcn_wmma_f32_16x16x32_f16(
            false, aq[f].v, false, bk.v, (short)0, s, false, false);
      }
      // scale 1/sqrt(64) then mask with (mask | eye)
#pragma unroll
      for (int r = 0; r < 8; ++r) {
        int q = q0 + r + 8 * half;
        int k = s0 + jj * 16 + ln;
        int mk = mkRow[(size_t)q * SLEN + k];
        s[r] = (mk | (q == k)) ? (s[r] * 0.125f) : -1.0e9f;
      }
      sc[jj] = s;
    }

    // ---- online softmax: joint row max over the 32 new columns ----
    float cf[8], mnew[8];
#pragma unroll
    for (int r = 0; r < 8; ++r) {
      float v = fmaxf(sc[0][r], sc[1][r]);
#pragma unroll
      for (int off = 1; off < 16; off <<= 1)
        v = fmaxf(v, __shfl_xor(v, off, 16));
      mnew[r] = fmaxf(mrun[r], v);
      cf[r]   = __expf(mrun[r] - mnew[r]);
      mrun[r] = mnew[r];
    }

    // p = exp(s - m); denominator uses p, numerator uses p*group_prob
#pragma unroll
    for (int jj = 0; jj < 2; ++jj)
#pragma unroll
      for (int r = 0; r < 8; ++r) {
        int q = q0 + r + 8 * half;
        int k = s0 + jj * 16 + ln;
        float p = __expf(sc[jj][r] - mnew[r]);
        sc[jj][r] = p;
        float g = gpRow[(size_t)q * SLEN + k];
        Pw[(r + 8 * half) * PSTR + jj * 16 + ln] = (_Float16)(p * g);
      }
#pragma unroll
    for (int r = 0; r < 8; ++r) {
      float v = sc[0][r] + sc[1][r];
#pragma unroll
      for (int off = 1; off < 16; off <<= 1)
        v += __shfl_xor(v, off, 16);
      lsum[r] = lsum[r] * cf[r] + v;
    }
    // rescale output accumulators
#pragma unroll
    for (int j2 = 0; j2 < 4; ++j2)
#pragma unroll
      for (int r = 0; r < 8; ++r)
        acc[j2][r] *= cf[r];

    // re-layout P (C-layout in LDS) into an A-fragment; same-wave DS is in-order
    asm volatile("" ::: "memory");
    AFrag ap;
#pragma unroll
    for (int j = 0; j < 8; ++j) {
      int ko = apair(j, half);
      ap.v[2 * j]     = Pw[ln * PSTR + ko];
      ap.v[2 * j + 1] = Pw[ln * PSTR + ko + 1];
    }

    // ---- P x V: 4 WMMA over the 64 output dims ----
#pragma unroll
    for (int j2 = 0; j2 < 4; ++j2) {
      BFrag bv;
#pragma unroll
      for (int e = 0; e < 16; ++e)
        bv.v[e] = vBase[(size_t)(s0 + apair(e >> 1, half) + (e & 1)) * DK + j2 * 16 + ln];
      acc[j2] = __builtin_amdgcn_wmma_f32_16x16x32_f16(
          false, ap.v, false, bv.v, (short)0, acc[j2], false, false);
    }
  }

  // write x[b, q, h*64+d] as f16 (merged-head layout for the output GEMM)
  float invl[8];
#pragma unroll
  for (int r = 0; r < 8; ++r) invl[r] = 1.0f / lsum[r];
#pragma unroll
  for (int j2 = 0; j2 < 4; ++j2)
#pragma unroll
    for (int r = 0; r < 8; ++r) {
      int q   = q0 + r + 8 * half;
      int col = h * DK + j2 * 16 + ln;
      Xo[((size_t)b * SLEN + q) * DMODEL + col] = (_Float16)(acc[j2][r] * invl[r]);
    }
}

// ---------------------------------------------------------------------------
// Kernel 3: output projection  out = X @ Wh^T + bh  (f16 in, f32 out)
// A tile is a pure f16 copy -> use async global->LDS DMA (ASYNCcnt-tracked).
// ---------------------------------------------------------------------------
__global__ __launch_bounds__(256) void outproj_kernel(
    const _Float16* __restrict__ X, const float* __restrict__ W,
    const float* __restrict__ bias, float* __restrict__ Out)
{
  __shared__ _Float16 As[128 * ASTR];
  __shared__ _Float16 Bs[32 * BSTR];
  const int tid  = threadIdx.x;
  const int lane = tid & 31;
  const int wave = tid >> 5;
  const int half = lane >> 4;
  const int ln   = lane & 15;
  const int wr   = wave & 3;
  const int wc   = wave >> 2;
  const int row0 = blockIdx.x * 128;
  const int col0 = blockIdx.y * 64;

  v8f acc[2][2] = {};

  for (int kc = 0; kc < DMODEL; kc += 32) {
    // stage A via async DMA: 512 x 16B chunks, 2 per thread
#pragma unroll
    for (int i = 0; i < 2; ++i) {
      int idx = tid + i * 256;            // 0..511
      int r   = idx >> 2;                 // 0..127
      int c8  = (idx & 3) * 8;            // 0,8,16,24
      uint32_t lds = (uint32_t)(uintptr_t)&As[r * ASTR + c8];
      uint64_t ga  = (uint64_t)(uintptr_t)&X[(size_t)(row0 + r) * DMODEL + kc + c8];
      asm volatile("global_load_async_to_lds_b128 %0, %1, off"
                   :: "v"(lds), "v"(ga) : "memory");
    }
    // stage B transposed with f32->f16 conversion, packed n-pairs -> b32 stores
#pragma unroll
    for (int i = 0; i < 4; ++i) {
      int idx = tid + i * 256;            // n-pair index 0..1023
      int np = idx >> 5, c = idx & 31;
      int n = np * 2;
      PackH pk;
      pk.h[0] = (_Float16)W[(size_t)(col0 + n)     * DMODEL + kc + c];
      pk.h[1] = (_Float16)W[(size_t)(col0 + n + 1) * DMODEL + kc + c];
      *(uint32_t*)(void*)&Bs[c * BSTR + n] = pk.u;
    }
    asm volatile("s_wait_asynccnt 0x0" ::: "memory");
    __syncthreads();

    AFrag a[2];
    BFrag bf[2];
#pragma unroll
    for (int i = 0; i < 2; ++i) {
      int m = wr * 32 + i * 16 + ln;
#pragma unroll
      for (int j = 0; j < 8; ++j)
        a[i].u[j] = *(const uint32_t*)(const void*)&As[m * ASTR + apair(j, half)];
    }
#pragma unroll
    for (int j = 0; j < 2; ++j) {
#pragma unroll
      for (int e = 0; e < 16; ++e)
        bf[j].v[e] = Bs[(apair(e >> 1, half) + (e & 1)) * BSTR + wc * 32 + j * 16 + ln];
    }
#pragma unroll
    for (int i = 0; i < 2; ++i)
#pragma unroll
      for (int j = 0; j < 2; ++j)
        acc[i][j] = __builtin_amdgcn_wmma_f32_16x16x32_f16(
            false, a[i].v, false, bf[j].v, (short)0, acc[i][j], false, false);
    __syncthreads();
  }

#pragma unroll
  for (int i = 0; i < 2; ++i)
#pragma unroll
    for (int j = 0; j < 2; ++j)
#pragma unroll
      for (int r = 0; r < 8; ++r) {
        int row = row0 + wr * 32 + i * 16 + r + 8 * half;
        int col = col0 + wc * 32 + j * 16 + ln;
        Out[(size_t)row * DMODEL + col] = acc[i][j][r] + bias[col];
      }
}

// ---------------------------------------------------------------------------
extern "C" void kernel_launch(void* const* d_in, const int* in_sizes, int n_in,
                              void* d_out, int out_size, void* d_ws, size_t ws_size,
                              hipStream_t stream) {
  const float* query = (const float*)d_in[0];
  const float* key   = (const float*)d_in[1];
  const float* value = (const float*)d_in[2];
  const float* gp    = (const float*)d_in[3];
  const float* Wq    = (const float*)d_in[4];
  const float* Wk    = (const float*)d_in[5];
  const float* Wv    = (const float*)d_in[6];
  const float* Wh    = (const float*)d_in[7];
  const float* bh    = (const float*)d_in[8];
  const int*   mask  = (const int*)d_in[9];

  const size_t elems = (size_t)BATCH * NHEAD * SLEN * DK; // 4M f16 per tensor
  _Float16* Qp = (_Float16*)d_ws;
  _Float16* Kt = Qp + elems;
  _Float16* Vp = Kt + elems;
  _Float16* Xo = Vp + elems;   // total 32 MB of workspace

  proj_qkv_kernel<<<dim3(64, 8, 3), 256, 0, stream>>>(query, key, value,
                                                      Wq, Wk, Wv, Qp, Kt, Vp);
  attn_kernel<<<dim3((BATCH * NHEAD * (SLEN / 16)) / 8), 256, 0, stream>>>(
      Qp, Kt, Vp, gp, mask, Xo);
  outproj_kernel<<<dim3(64, 8), 256, 0, stream>>>(Xo, Wh, bh, (float*)d_out);
}
